// DecoderRNN_39256001085765
// MI455X (gfx1250) — compile-verified, matrix operand order checked by reading, not run on hardware
//
#include <hip/hip_runtime.h>

// ---------------------------------------------------------------------------
// DecoderRNN single step for MI455X (gfx1250, wave32).
// Attention energy GEMM: LDS-staged A panel + 2Mx4N register blocking on
// v_wmma_f32_16x16x32_bf16 (0.5KB global traffic per wmma). All GEMVs are
// bandwidth-optimal float4 loads (one row per wave32, shuffle reduction).
// ---------------------------------------------------------------------------

#define H   2048
#define S   512
#define VSZ 50257

typedef __attribute__((ext_vector_type(16))) __bf16 v16bf;
typedef __attribute__((ext_vector_type(8)))  float  v8f;

__device__ __forceinline__ unsigned short f2bf(float f) {
    unsigned int u = __float_as_uint(f);
    u += 0x7fffu + ((u >> 16) & 1u);        // round-to-nearest-even
    return (unsigned short)(u >> 16);
}

// ---------------- fp32 -> bf16 conversion of W2 = attn_W[:,H:] -------------
__global__ __launch_bounds__(256) void cvt_w2_kernel(const float* __restrict__ attnW,
                                                     unsigned short* __restrict__ w2b) {
    int i = blockIdx.x * 256 + threadIdx.x;      // H*H elements
    int row = i >> 11, col = i & (H - 1);
    w2b[i] = f2bf(attnW[(size_t)row * (2 * H) + H + col]);
}

// ---------------- generic GEMV: y[row] = dot(W[row,:K], x) + b[row] --------
__global__ __launch_bounds__(256) void gemv_kernel(const float* __restrict__ W,
                                                   const float* __restrict__ x,
                                                   const float* __restrict__ bias,
                                                   float* __restrict__ y,
                                                   int nrows, int K, int rowStride) {
    int wave = threadIdx.x >> 5, lane = threadIdx.x & 31;
    int row = blockIdx.x * 8 + wave;
    if (row >= nrows) return;
    const float* w = W + (size_t)row * rowStride;
    float s = 0.f;
    for (int k = lane * 4; k < K; k += 128) {
        float4 wv = *(const float4*)(w + k);
        float4 xv = *(const float4*)(x + k);
        s += wv.x * xv.x + wv.y * xv.y + wv.z * xv.z + wv.w * xv.w;
    }
    for (int m = 16; m >= 1; m >>= 1) s += __shfl_xor(s, m);
    if (lane == 0) y[row] = s + bias[row];
}

// ---------------- attention energies via WMMA ------------------------------
// scores[s] = sum_j tanh(u[j] + enc[s,:]·W2[j,:]) * v[j]
// Grid: 16 blocks, one per 32-row s-strip. A strip (32x2048 bf16 = 128KB) is
// staged in LDS (fp32->bf16 fused). Each of 8 waves owns 16 j-tiles, processed
// in 4 groups of 4 with 2Mx4N register blocking: per k-step 2 LDS A-frags +
// 4 global B-frags feed 8 v_wmma_f32_16x16x32_bf16.
__global__ __launch_bounds__(256) void attn_wmma_kernel(const float* __restrict__ enc,
                                                        const unsigned short* __restrict__ w2b,
                                                        const float* __restrict__ u,
                                                        const float* __restrict__ vvec,
                                                        float* __restrict__ scores) {
    extern __shared__ __align__(32) unsigned short As[];   // [32][H] bf16
    __shared__ float wave_sums[8][32];

    const int s0 = blockIdx.x * 32;

    // ---- cooperative staging: enc[s0:s0+32, :] fp32 -> bf16 in LDS --------
    {
        const float4* src = (const float4*)(enc + (size_t)s0 * H);
        const int nvec = (32 * H) / 4;                     // 16384 float4s
        for (int idx = threadIdx.x; idx < nvec; idx += 256) {
            float4 f = src[idx];
            ushort4 o;
            o.x = f2bf(f.x); o.y = f2bf(f.y); o.z = f2bf(f.z); o.w = f2bf(f.w);
            *(ushort4*)(As + (size_t)idx * 4) = o;
        }
    }
    __syncthreads();

    const int lane  = threadIdx.x & 31;
    const int wave  = threadIdx.x >> 5;          // 0..7
    const int mrow  = lane & 15;                 // A row / B column per lane
    const int khalf = (lane >> 4) * 16;          // K half select

    const unsigned short* a0p = As + (size_t)mrow * H + khalf;        // M-tile 0
    const unsigned short* a1p = As + (size_t)(16 + mrow) * H + khalf; // M-tile 1

    float psum[2][8];
#pragma unroll
    for (int mi = 0; mi < 2; ++mi)
#pragma unroll
        for (int r = 0; r < 8; ++r) psum[mi][r] = 0.f;

    for (int g = 0; g < 4; ++g) {
        const int jbase = (wave * 16 + g * 4) * 16;        // first j of group
        const unsigned short* bp0 = w2b + (size_t)(jbase +  0 + mrow) * H + khalf;
        const unsigned short* bp1 = w2b + (size_t)(jbase + 16 + mrow) * H + khalf;
        const unsigned short* bp2 = w2b + (size_t)(jbase + 32 + mrow) * H + khalf;
        const unsigned short* bp3 = w2b + (size_t)(jbase + 48 + mrow) * H + khalf;

        v8f acc[2][4];
#pragma unroll
        for (int mi = 0; mi < 2; ++mi)
#pragma unroll
            for (int i = 0; i < 4; ++i) acc[mi][i] = (v8f){};

        for (int k0 = 0; k0 < H; k0 += 32) {
            v16bf a0 = *(const v16bf*)(a0p + k0);          // 32B from LDS
            v16bf a1 = *(const v16bf*)(a1p + k0);
            v16bf b0 = *(const v16bf*)(bp0 + k0);          // 32B from global
            v16bf b1 = *(const v16bf*)(bp1 + k0);
            v16bf b2 = *(const v16bf*)(bp2 + k0);
            v16bf b3 = *(const v16bf*)(bp3 + k0);
            acc[0][0] = __builtin_amdgcn_wmma_f32_16x16x32_bf16(false, a0, false, b0, (short)0, acc[0][0], false, false);
            acc[1][0] = __builtin_amdgcn_wmma_f32_16x16x32_bf16(false, a1, false, b0, (short)0, acc[1][0], false, false);
            acc[0][1] = __builtin_amdgcn_wmma_f32_16x16x32_bf16(false, a0, false, b1, (short)0, acc[0][1], false, false);
            acc[1][1] = __builtin_amdgcn_wmma_f32_16x16x32_bf16(false, a1, false, b1, (short)0, acc[1][1], false, false);
            acc[0][2] = __builtin_amdgcn_wmma_f32_16x16x32_bf16(false, a0, false, b2, (short)0, acc[0][2], false, false);
            acc[1][2] = __builtin_amdgcn_wmma_f32_16x16x32_bf16(false, a1, false, b2, (short)0, acc[1][2], false, false);
            acc[0][3] = __builtin_amdgcn_wmma_f32_16x16x32_bf16(false, a0, false, b3, (short)0, acc[0][3], false, false);
            acc[1][3] = __builtin_amdgcn_wmma_f32_16x16x32_bf16(false, a1, false, b3, (short)0, acc[1][3], false, false);
        }

        // fused epilogue: energies -> tanh -> dot with v (per-lane partials)
#pragma unroll
        for (int i = 0; i < 4; ++i) {
            const int j = jbase + i * 16 + (lane & 15);
            const float uj = u[j];
            const float vj = vvec[j];
#pragma unroll
            for (int r = 0; r < 8; ++r) {
                psum[0][r] += tanhf(acc[0][i][r] + uj) * vj;
                psum[1][r] += tanhf(acc[1][i][r] + uj) * vj;
            }
        }
    }

    // reduce over the 16 N-lanes of each half (masks stay in 16-lane groups)
#pragma unroll
    for (int mi = 0; mi < 2; ++mi)
#pragma unroll
        for (int r = 0; r < 8; ++r) {
            float x = psum[mi][r];
            x += __shfl_xor(x, 1);
            x += __shfl_xor(x, 2);
            x += __shfl_xor(x, 4);
            x += __shfl_xor(x, 8);
            psum[mi][r] = x;
        }
    if ((lane & 15) == 0) {
        const int h = lane >> 4;                 // rows r + 8h within tile
#pragma unroll
        for (int mi = 0; mi < 2; ++mi)
#pragma unroll
            for (int r = 0; r < 8; ++r)
                wave_sums[wave][mi * 16 + r + 8 * h] = psum[mi][r];
    }
    __syncthreads();
    if (threadIdx.x < 32) {                      // deterministic fixed-order sum
        float s = 0.f;
        for (int w = 0; w < 8; ++w) s += wave_sums[w][threadIdx.x];
        scores[s0 + threadIdx.x] = s;
    }
}

// ---------------- softmax over S=512 (single block) ------------------------
__global__ __launch_bounds__(512) void softmax_kernel(const float* __restrict__ scores,
                                                      float* __restrict__ attnw,
                                                      float* __restrict__ out_attn) {
    __shared__ float red[512];
    int t = threadIdx.x;
    float x = scores[t];
    red[t] = x; __syncthreads();
    for (int s = 256; s > 0; s >>= 1) {
        if (t < s) red[t] = fmaxf(red[t], red[t + s]);
        __syncthreads();
    }
    float m = red[0]; __syncthreads();
    float e = expf(x - m);
    red[t] = e; __syncthreads();
    for (int s = 256; s > 0; s >>= 1) {
        if (t < s) red[t] += red[t + s];
        __syncthreads();
    }
    float w = e / red[0];
    attnw[t] = w;
    out_attn[t] = w;
}

// ---------------- context[j] = sum_s attnw[s] * enc[s,j] -------------------
__global__ __launch_bounds__(256) void context_kernel(const float* __restrict__ enc,
                                                      const float* __restrict__ attnw,
                                                      float* __restrict__ context) {
    __shared__ float aw[S];
    for (int i = threadIdx.x; i < S; i += 256) aw[i] = attnw[i];
    __syncthreads();
    int j = blockIdx.x * 256 + threadIdx.x;          // H outputs, coalesced
    float s = 0.f;
    for (int t = 0; t < S; ++t) s += aw[t] * enc[(size_t)t * H + j];
    context[j] = s;
}

// ---------------- cat = [E[tok], context] ----------------------------------
__global__ __launch_bounds__(256) void build_cat_kernel(const float* __restrict__ E,
                                                        const int* __restrict__ tok,
                                                        const float* __restrict__ context,
                                                        float* __restrict__ cat) {
    int i = blockIdx.x * 256 + threadIdx.x;          // 2H elements
    size_t t = (size_t)tok[0];
    cat[i] = (i < H) ? E[t * H + i] : context[i - H];
}

// ---------------- GRU gates -------------------------------------------------
__global__ __launch_bounds__(256) void gru_gate_kernel(const float* __restrict__ gi,
                                                       const float* __restrict__ gh,
                                                       const float* __restrict__ h,
                                                       float* __restrict__ out_h) {
    int i = blockIdx.x * 256 + threadIdx.x;          // H elements
    float r = 1.f / (1.f + expf(-(gi[i] + gh[i])));
    float z = 1.f / (1.f + expf(-(gi[H + i] + gh[H + i])));
    float n = tanhf(gi[2 * H + i] + r * gh[2 * H + i]);
    out_h[i] = (1.f - z) * n + z * h[i];
}

// ---------------- vocab logits + per-block max (412 MB stream) -------------
__global__ __launch_bounds__(256) void logits_kernel(const float* __restrict__ outW,
                                                     const float* __restrict__ h,
                                                     const float* __restrict__ outb,
                                                     float* __restrict__ logits,
                                                     float* __restrict__ blockmax) {
    __shared__ float hs[H];
    for (int i = threadIdx.x; i < H; i += 256) hs[i] = h[i];
    __syncthreads();
    int wave = threadIdx.x >> 5, lane = threadIdx.x & 31;
    int row = blockIdx.x * 8 + wave;
    float val = -3.0e38f;
    if (row < VSZ) {
        const float* w = outW + (size_t)row * H;
        float s = 0.f;
        for (int k = lane * 4; k < H; k += 128) {
            if (k + 1024 < H) __builtin_prefetch(w + k + 1024, 0, 1);
            float4 wv = *(const float4*)(w + k);
            float4 xv = *(const float4*)(hs + k);
            s += wv.x * xv.x + wv.y * xv.y + wv.z * xv.z + wv.w * xv.w;
        }
        for (int m = 16; m >= 1; m >>= 1) s += __shfl_xor(s, m);
        s += outb[row];
        if (lane == 0) logits[row] = s;
        val = s;
    }
    __shared__ float wmax[8];
    if (lane == 0) wmax[wave] = val;
    __syncthreads();
    if (threadIdx.x == 0) {
        float m = wmax[0];
        for (int w = 1; w < 8; ++w) m = fmaxf(m, wmax[w]);
        blockmax[blockIdx.x] = m;
    }
}

// ---------------- global max + logZ (single block) -------------------------
__global__ __launch_bounds__(1024) void lsm_reduce_kernel(const float* __restrict__ logits,
                                                          const float* __restrict__ blockmax,
                                                          int nblocks,
                                                          float* __restrict__ scalars) {
    __shared__ float red[1024];
    float m = -3.0e38f;
    for (int i = threadIdx.x; i < nblocks; i += 1024) m = fmaxf(m, blockmax[i]);
    red[threadIdx.x] = m; __syncthreads();
    for (int s = 512; s > 0; s >>= 1) {
        if (threadIdx.x < s) red[threadIdx.x] = fmaxf(red[threadIdx.x], red[threadIdx.x + s]);
        __syncthreads();
    }
    float gmax = red[0]; __syncthreads();
    float sum = 0.f;
    for (int i = threadIdx.x; i < VSZ; i += 1024) sum += expf(logits[i] - gmax);
    red[threadIdx.x] = sum; __syncthreads();
    for (int s = 512; s > 0; s >>= 1) {
        if (threadIdx.x < s) red[threadIdx.x] += red[threadIdx.x + s];
        __syncthreads();
    }
    if (threadIdx.x == 0) { scalars[0] = gmax; scalars[1] = logf(red[0]); }
}

__global__ __launch_bounds__(256) void lsm_write_kernel(const float* __restrict__ logits,
                                                        const float* __restrict__ scalars,
                                                        float* __restrict__ out) {
    int i = blockIdx.x * 256 + threadIdx.x;
    if (i < VSZ) out[i] = logits[i] - scalars[0] - scalars[1];
}

// ---------------------------------------------------------------------------
extern "C" void kernel_launch(void* const* d_in, const int* in_sizes, int n_in,
                              void* d_out, int out_size, void* d_ws, size_t ws_size,
                              hipStream_t stream) {
    (void)in_sizes; (void)n_in; (void)out_size; (void)ws_size;

    const int*   tok    = (const int*)d_in[0];
    const float* hidden = (const float*)d_in[1];
    const float* enc    = (const float*)d_in[2];
    const float* E      = (const float*)d_in[3];
    const float* attn_W = (const float*)d_in[4];
    const float* attn_b = (const float*)d_in[5];
    const float* vvec   = (const float*)d_in[6];
    const float* fc_W   = (const float*)d_in[7];
    const float* fc_b   = (const float*)d_in[8];
    const float* W_ih   = (const float*)d_in[9];
    const float* W_hh   = (const float*)d_in[10];
    const float* b_ih   = (const float*)d_in[11];
    const float* b_hh   = (const float*)d_in[12];
    const float* out_W  = (const float*)d_in[13];
    const float* out_b  = (const float*)d_in[14];

    float* out       = (float*)d_out;
    float* out_logp  = out;                  // VSZ
    float* out_h     = out + VSZ;            // H
    float* out_attn  = out + VSZ + H;        // S

    // workspace layout (256B-aligned bump allocator)
    char*  base = (char*)d_ws;
    size_t off  = 0;
    auto alloc = [&](size_t bytes) {
        void* p = base + off;
        off = (off + bytes + 255) & ~(size_t)255;
        return p;
    };
    unsigned short* w2b = (unsigned short*)alloc((size_t)H * H * 2);
    float* u        = (float*)alloc(H * 4);
    float* scores   = (float*)alloc(S * 4);
    float* attnw    = (float*)alloc(S * 4);
    float* context  = (float*)alloc(H * 4);
    float* cat      = (float*)alloc(2 * H * 4);
    float* xbuf     = (float*)alloc(H * 4);
    float* gi       = (float*)alloc(3 * H * 4);
    float* gh       = (float*)alloc(3 * H * 4);
    float* logits   = (float*)alloc((size_t)VSZ * 4);
    const int nLogitBlocks = (VSZ + 7) / 8;
    float* blockmax = (float*)alloc((size_t)nLogitBlocks * 4);
    float* scalars  = (float*)alloc(2 * 4);

    // 1) bf16 conversion of the streamed B operand (halves L2 traffic)
    cvt_w2_kernel<<<(H * H) / 256, 256, 0, stream>>>(attn_W, w2b);

    // 2) u[j] = h . attn_W[j, :H] + attn_b[j]
    gemv_kernel<<<H / 8, 256, 0, stream>>>(attn_W, hidden, attn_b, u, H, H, 2 * H);

    // 3) attention scores via WMMA (LDS-staged A, 2Mx4N register blocking)
    attn_wmma_kernel<<<S / 32, 256, 32 * H * sizeof(unsigned short), stream>>>(
        enc, w2b, u, vvec, scores);

    // 4) softmax over S
    softmax_kernel<<<1, 512, 0, stream>>>(scores, attnw, out_attn);

    // 5) context = attnw @ enc
    context_kernel<<<H / 256, 256, 0, stream>>>(enc, attnw, context);

    // 6) cat = [E[tok], context];  x = fc(cat)
    build_cat_kernel<<<(2 * H) / 256, 256, 0, stream>>>(E, tok, context, cat);
    gemv_kernel<<<H / 8, 256, 0, stream>>>(fc_W, cat, fc_b, xbuf, H, 2 * H, 2 * H);

    // 7) GRU gates
    gemv_kernel<<<(3 * H) / 8, 256, 0, stream>>>(W_ih, xbuf, b_ih, gi, 3 * H, H, H);
    gemv_kernel<<<(3 * H) / 8, 256, 0, stream>>>(W_hh, hidden, b_hh, gh, 3 * H, H, H);
    gru_gate_kernel<<<H / 256, 256, 0, stream>>>(gi, gh, hidden, out_h);

    // 8) vocab logits (dominant 412MB stream) + log-softmax
    logits_kernel<<<nLogitBlocks, 256, 0, stream>>>(out_W, out_h, out_b, logits, blockmax);
    lsm_reduce_kernel<<<1, 1024, 0, stream>>>(logits, blockmax, nLogitBlocks, scalars);
    lsm_write_kernel<<<(VSZ + 255) / 256, 256, 0, stream>>>(logits, scalars, out_logp);
}
